// Reservoir_5669356833869
// MI455X (gfx1250) — compile-verified, matrix operand order checked by reading, not run on hardware
//
#include <hip/hip_runtime.h>
#include <hip/hip_bf16.h>
#include <stdint.h>

#define T_STEPS 2048
#define BATCH   32
#define DIN     256
#define HDIM    512
#define LEAK    0.9f

typedef __attribute__((ext_vector_type(16))) __bf16          v16bf;
typedef __attribute__((ext_vector_type(8)))  __bf16          v8bf;
typedef __attribute__((ext_vector_type(8)))  float           v8f;
typedef __attribute__((ext_vector_type(16))) unsigned short  us16;
typedef __attribute__((ext_vector_type(8)))  unsigned short  us8;

union BfPack16 { us16 u; v16bf b; };
union BfPack8  { us8  u; v8bf  b; };

__device__ __forceinline__ unsigned short bfbits(float f) {
  union { __bf16 b; unsigned short u; } v; v.b = (__bf16)f; return v.u;
}

// tanh via v_exp2/v_rcp (trans ops co-execute with WMMA on CDNA5)
__device__ __forceinline__ float fast_tanh(float x) {
  float ax = __builtin_fabsf(x);
  float e  = __builtin_amdgcn_exp2f(ax * -2.8853900817779268f); // exp(-2|x|)
  float r  = (1.0f - e) * __builtin_amdgcn_rcpf(1.0f + e);
  return x < 0.0f ? -r : r;
}

// ---------------------------------------------------------------------------
// Phase A: U = A @ W^T + b   (A: [M=T*B, lda] fp32, W: [H][K] fp32)
// U written bf16 in WMMA-C tile order: [tileM][tileN][lane][8] so the
// recurrence loads each C fragment with one global_load_b128.
// Each wave computes one 16x16 tile; block = 8 waves = 32 rows x 64 cols.
// ---------------------------------------------------------------------------
__global__ __launch_bounds__(256) void uproj_kernel(
    const float* __restrict__ A, int lda, int K,
    const float* __restrict__ W,
    const float* __restrict__ bias,
    unsigned short* __restrict__ U) {
  const int tid  = threadIdx.x;
  const int wave = tid >> 5, lane = tid & 31;
  const int lh = lane & 15, lq = lane >> 4;
  const int tileM = blockIdx.x * 2 + (wave >> 2);
  const int tileN = blockIdx.y * 4 + (wave & 3);

  const float* arow = A + (size_t)(tileM * 16 + lh) * lda;
  const float* wrow = W + (size_t)(tileN * 16 + lh) * K;

  v8f acc = {};
  for (int k0 = 0; k0 < K; k0 += 32) {
    const float* ap = arow + k0 + lq * 16;
    const float* wp = wrow + k0 + lq * 16;
    v16bf a, b;
#pragma unroll
    for (int e = 0; e < 16; ++e) { a[e] = (__bf16)ap[e]; b[e] = (__bf16)wp[e]; }
    acc = __builtin_amdgcn_wmma_f32_16x16x32_bf16(
        false, a, false, b, (short)0, acc, false, false);
  }
  const float bv = bias[tileN * 16 + lh];
  BfPack8 o;
#pragma unroll
  for (int r = 0; r < 8; ++r) o.b[r] = (__bf16)(acc[r] + bv);
  us8* up = (us8*)(U + (((size_t)tileM * (HDIM / 16) + tileN) * 256) + lane * 8);
  *up = o.u;
}

// ---------------------------------------------------------------------------
// Phase B: sequential recurrence, single persistent workgroup (32 wave32s).
// h = (1-LEAK)*h + LEAK*tanh(U[t] + h @ w_hh^T)
//  - w_hh register-resident bf16 (128 VGPRs/wave)
//  - h kept in registers for the leak term (lane<->C-fragment map is static);
//    LDS ping-pong (2x32x512 bf16 = 64KB, chunk-swizzled) only for the
//    cross-wave A-operand exchange
//  - U[t+1] fragments software-pipelined under the WMMA chain
// ---------------------------------------------------------------------------
__device__ __forceinline__ int hidx(int row, int col) {
  int ch = ((col >> 4) ^ (row & 31)) & 31;      // swizzle 16-elem chunks
  return row * HDIM + ch * 16 + (col & 15);
}

__global__ __launch_bounds__(1024) void recur_kernel(
    const unsigned short* __restrict__ U,       // bf16, tile order (see above)
    const float* __restrict__ w_hh,             // [H][H] fp32
    float* __restrict__ out,                    // [T][B][2H] fp32 (+col_off)
    float* __restrict__ nexth,                  // [B][H] fp32
    int col_off) {
  __shared__ __align__(32) unsigned short hbuf[2][BATCH * HDIM];

  const int tid  = threadIdx.x;
  const int wave = tid >> 5, lane = tid & 31;
  const int lh = lane & 15, lq = lane >> 4;

  // Preload this wave's N-column block of w_hh as WMMA B operand, bf16.
  v16bf breg[16];
  {
    const float* wrow = w_hh + (size_t)(wave * 16 + lh) * HDIM;
#pragma unroll
    for (int kc = 0; kc < 16; ++kc) {
      const float* p = wrow + kc * 32 + lq * 16;
      v16bf t;
#pragma unroll
      for (int e = 0; e < 16; ++e) t[e] = (__bf16)p[e];
      breg[kc] = t;
    }
  }
  // h0 = 0 (LDS copy for A-operand; register copy for leak term)
  for (int i = tid; i < BATCH * HDIM; i += 1024) hbuf[0][i] = 0;
  v8f hreg0 = {}, hreg1 = {};
  __syncthreads();

  const int colg = wave * 16 + lh;
  const int NT = HDIM / 16;                      // 32 N-tiles
  const size_t fragOff = (size_t)wave * 256 + lane * 8;

  // prime the U pipeline with step 0 fragments
  BfPack8 uc0, uc1;
  uc0.u = *(const us8*)(U + fragOff);
  uc1.u = *(const us8*)(U + (size_t)NT * 256 + fragOff);

  int cur = 0;
#pragma unroll 1
  for (int t = 0; t < T_STEPS; ++t) {
    // accumulators start at input projection U[t] (bias included)
    v8f acc0, acc1;
#pragma unroll
    for (int r = 0; r < 8; ++r) { acc0[r] = (float)uc0.b[r]; acc1[r] = (float)uc1.b[r]; }

    // issue next step's fragment loads; latency hides under the WMMA chain
    if (t + 1 < T_STEPS) {
      const unsigned short* Un = U + (size_t)(2 * (t + 1)) * NT * 256;
      uc0.u = *(const us8*)(Un + fragOff);
      uc1.u = *(const us8*)(Un + (size_t)NT * 256 + fragOff);
      __builtin_prefetch(U + (size_t)(2 * (t + 2)) * NT * 256 + tid * 16, 0, 0);
    }

    const unsigned short* hb = hbuf[cur];
#pragma unroll
    for (int kc = 0; kc < 16; ++kc) {
      int kbase = kc * 32 + lq * 16;
      BfPack16 a0, a1;
      a0.u = *(const us16*)&hb[hidx(lh, kbase)];        // M-tile 0 rows
      a1.u = *(const us16*)&hb[hidx(16 + lh, kbase)];   // M-tile 1 rows
      acc0 = __builtin_amdgcn_wmma_f32_16x16x32_bf16(
          false, a0.b, false, breg[kc], (short)0, acc0, false, false);
      acc1 = __builtin_amdgcn_wmma_f32_16x16x32_bf16(
          false, a1.b, false, breg[kc], (short)0, acc1, false, false);
    }

    const int nxt = cur ^ 1;
    float* orow = out + (size_t)t * BATCH * (2 * HDIM) + col_off + colg;
#pragma unroll
    for (int r = 0; r < 8; ++r) {
      int m0 = lq * 8 + r, m1 = m0 + 16;
      float hn0 = (1.0f - LEAK) * hreg0[r] + LEAK * fast_tanh(acc0[r]);
      float hn1 = (1.0f - LEAK) * hreg1[r] + LEAK * fast_tanh(acc1[r]);
      hreg0[r] = hn0; hreg1[r] = hn1;
      orow[(size_t)m0 * (2 * HDIM)] = hn0;
      orow[(size_t)m1 * (2 * HDIM)] = hn1;
      hbuf[nxt][hidx(m0, colg)] = bfbits(hn0);
      hbuf[nxt][hidx(m1, colg)] = bfbits(hn1);
    }
    __syncthreads();
    cur = nxt;
  }

  // final hidden state from the register-resident h
#pragma unroll
  for (int r = 0; r < 8; ++r) {
    int m0 = lq * 8 + r, m1 = m0 + 16;
    nexth[(size_t)m0 * HDIM + colg] = hreg0[r];
    nexth[(size_t)m1 * HDIM + colg] = hreg1[r];
  }
}

// ---------------------------------------------------------------------------
extern "C" void kernel_launch(void* const* d_in, const int* in_sizes, int n_in,
                              void* d_out, int out_size, void* d_ws, size_t ws_size,
                              hipStream_t stream) {
  (void)in_sizes; (void)n_in; (void)out_size; (void)ws_size;
  const float* x    = (const float*)d_in[0];   // [T,B,Din]
  const float* wih0 = (const float*)d_in[1];   // [H,Din]
  const float* whh0 = (const float*)d_in[2];   // [H,H]
  const float* bih0 = (const float*)d_in[3];   // [H]
  const float* wih1 = (const float*)d_in[4];   // [H,H]
  const float* whh1 = (const float*)d_in[5];   // [H,H]
  const float* bih1 = (const float*)d_in[6];   // [H]

  float* out = (float*)d_out;                  // [T,B,2H] then [2,B,H]
  unsigned short* Ubuf = (unsigned short*)d_ws; // T*B*H bf16 = 64MB scratch

  float* nexth0 = out + (size_t)T_STEPS * BATCH * (2 * HDIM);
  float* nexth1 = nexth0 + (size_t)BATCH * HDIM;

  dim3 gridU(T_STEPS * BATCH / 32, HDIM / 64);

  // Layer 0
  uproj_kernel<<<gridU, 256, 0, stream>>>(x, DIN, DIN, wih0, bih0, Ubuf);
  recur_kernel<<<dim3(1), dim3(1024), 0, stream>>>(Ubuf, whh0, out, nexth0, 0);
  // Layer 1: input = out0 = d_out cols [0,512), row stride 1024
  uproj_kernel<<<gridU, 256, 0, stream>>>(out, 2 * HDIM, HDIM, wih1, bih1, Ubuf);
  recur_kernel<<<dim3(1), dim3(1024), 0, stream>>>(Ubuf, whh1, out, nexth1, HDIM);
}
